// GraphSAGE_78288663871650
// MI455X (gfx1250) — matrix operand order repaired and checked by reading, not compiled, not run on hardware
//
#include <hip/hip_runtime.h>
#include <hip/hip_fp16.h>

typedef __attribute__((ext_vector_type(16))) _Float16 v16h;
typedef __attribute__((ext_vector_type(8)))  float    v8f;
typedef __attribute__((ext_vector_type(4)))  int      v4i;

// ---------------------------------------------------------------------------
// gfx1250 async global->LDS copy (16B per lane), with fallbacks.
// Probe result: builtin exists; param0 is int4* in global AS (non-const).
// ---------------------------------------------------------------------------
__device__ __forceinline__ void async_cp16(_Float16* lds, const _Float16* g) {
#if __has_builtin(__builtin_amdgcn_global_load_async_to_lds_b128)
  typedef __attribute__((address_space(1))) v4i GV4;
  typedef __attribute__((address_space(3))) v4i LV4;
  __builtin_amdgcn_global_load_async_to_lds_b128(
      (GV4*)g, (LV4*)lds, 0, 0);
#else
  unsigned lo = (unsigned)(unsigned long long)lds;  // LDS offset = addr[31:0]
  asm volatile("global_load_async_to_lds_b128 %0, %1, off"
               :: "v"(lo), "v"((unsigned long long)g) : "memory");
#endif
}

__device__ __forceinline__ void async_join() {
#if __has_builtin(__builtin_amdgcn_s_wait_asynccnt)
  __builtin_amdgcn_s_wait_asynccnt(0);
#else
  asm volatile("s_wait_asynccnt 0x0" ::: "memory");
#endif
}

// ---------------------------------------------------------------------------
// Utility kernels
// ---------------------------------------------------------------------------

__global__ void zero_f32(float* __restrict__ p, long long n) {
  long long i = (long long)blockIdx.x * blockDim.x + threadIdx.x;
  if (i < n) p[i] = 0.0f;
}

// h[i,:] = x[n_id[i],:]   (float4 vectorized, 16 float4 per row of 64)
__global__ void gather_rows(const float4* __restrict__ x,
                            const int* __restrict__ nid,
                            float4* __restrict__ h, long long n) {
  long long i = (long long)blockIdx.x * blockDim.x + threadIdx.x;
  if (i >= n * 16) return;
  long long row = i >> 4;
  int c = (int)(i & 15);
  long long s = nid[row];
  h[i] = x[s * 16 + c];
}

// cnt[dst[e]] += 1
__global__ void edge_count(const int* __restrict__ dst,
                           float* __restrict__ cnt, long long e) {
  long long i = (long long)blockIdx.x * blockDim.x + threadIdx.x;
  if (i < e) atomicAdd(&cnt[dst[i]], 1.0f);
}

// agg[dst[e],:] += h[src[e],:]   (float4 gather, 4 scalar f32 atomics)
__global__ void edge_scatter(const int* __restrict__ src,
                             const int* __restrict__ dst,
                             const float4* __restrict__ h,
                             float* __restrict__ agg, long long e) {
  long long i = (long long)blockIdx.x * blockDim.x + threadIdx.x;
  if (i >= e * 16) return;
  long long ed = i >> 4;
  int c = (int)(i & 15);
  long long s = src[ed];
  long long d = dst[ed];
  float4 v = h[s * 16 + c];
  float* b = agg + d * 64 + c * 4;
  atomicAdd(b + 0, v.x);
  atomicAdd(b + 1, v.y);
  atomicAdd(b + 2, v.z);
  atomicAdd(b + 3, v.w);
}

// ---------------------------------------------------------------------------
// Weight packing: fp32 W[64][outCols] -> f16 B-fragments for
// v_wmma_f32_16x16x32_f16.  Per ISA 7.12.2 (16-bit B 32x16, wave32):
//   column N = lane&15 ; K = kc + 8*(lane>>4) + e (e<8), kc + 16 + 8*(lane>>4) + (e-8) (e>=8).
// Fragment index: dst[((kcIdx*nTiles + t)*32 + lane)*16 + e]
// ---------------------------------------------------------------------------
__global__ void pack_weight(const float* __restrict__ W,
                            _Float16* __restrict__ dstp,
                            int outCols, int nTiles) {
  int tid = blockIdx.x * blockDim.x + threadIdx.x;
  int total = 2 * nTiles * 32 * 16;
  if (tid >= total) return;
  int e    = tid & 15;
  int lane = (tid >> 4) & 31;
  int rem  = tid >> 9;           // kcIdx*nTiles + t
  int t     = rem % nTiles;
  int kcIdx = rem / nTiles;
  int half = lane >> 4;
  int col  = t * 16 + (lane & 15);
  int k    = kcIdx * 32 + (e < 8 ? half * 8 + e : 16 + half * 8 + (e - 8));
  float v = (col < outCols) ? W[k * outCols + col] : 0.0f;
  dstp[tid] = (_Float16)v;
}

// ---------------------------------------------------------------------------
// A-fragment build: per ISA 7.12.2 (16-bit A 16x32, wave32):
//   row M = lane&15 ; half = lane>>4
//   e=0..7  -> K = kc + 8*half + e        (contiguous run of 8)
//   e=8..15 -> K = kc + 16 + 8*half + (e-8)
// Both runs are 32B aligned in a 64-float row -> two float4 loads each.
// ---------------------------------------------------------------------------
__device__ __forceinline__ v16h load_a_frag(const float* __restrict__ P,
                                            long long rowA, int kc, int half,
                                            float scale) {
  const float4* p4 = (const float4*)(P + rowA * 64 + kc + half * 8);
  float4 r0 = p4[0], r1 = p4[1];
  const float4* q4 = (const float4*)(P + rowA * 64 + kc + 16 + half * 8);
  float4 r2 = q4[0], r3 = q4[1];
  v16h a;
  a[0]  = (_Float16)(r0.x * scale); a[1]  = (_Float16)(r0.y * scale);
  a[2]  = (_Float16)(r0.z * scale); a[3]  = (_Float16)(r0.w * scale);
  a[4]  = (_Float16)(r1.x * scale); a[5]  = (_Float16)(r1.y * scale);
  a[6]  = (_Float16)(r1.z * scale); a[7]  = (_Float16)(r1.w * scale);
  a[8]  = (_Float16)(r2.x * scale); a[9]  = (_Float16)(r2.y * scale);
  a[10] = (_Float16)(r2.z * scale); a[11] = (_Float16)(r2.w * scale);
  a[12] = (_Float16)(r3.x * scale); a[13] = (_Float16)(r3.y * scale);
  a[14] = (_Float16)(r3.z * scale); a[15] = (_Float16)(r3.w * scale);
  return a;
}

// ---------------------------------------------------------------------------
// Fused SAGE GEMM: Out = act( Hs@Ws + (Agg/max(cnt,1))@Wn + bias )
// One wave per 16-row tile; NT output tiles of 16 cols; K=64 in two 32-chunks.
// Packed weights staged once per block into LDS via async global->LDS copies.
// ---------------------------------------------------------------------------
template <int NT, bool NEIGH, bool RELU>
__global__ __launch_bounds__(256) void sage_gemm(
    const float* __restrict__ Hs, const float* __restrict__ Agg,
    const float* __restrict__ Cnt,
    const _Float16* __restrict__ Bs, const _Float16* __restrict__ Bn,
    const float* __restrict__ bias, float* __restrict__ Out,
    int nRows, int outCols) {
  constexpr int FRAG_HALVES = NT * 2 * 512;  // per weight matrix
  __shared__ __attribute__((aligned(32))) _Float16 sBs[FRAG_HALVES];
  __shared__ __attribute__((aligned(32))) _Float16 sBn[NEIGH ? FRAG_HALVES : 16];

  // --- stage packed weights into LDS (16B per lane, ASYNCcnt tracked) ---
  {
    constexpr int CH = FRAG_HALVES / 8;  // number of 16B chunks
    for (int c = threadIdx.x; c < CH; c += 256) {
      async_cp16(&sBs[c * 8], Bs + c * 8);
      if constexpr (NEIGH) async_cp16(&sBn[c * 8], Bn + c * 8);
    }
    async_join();
    __syncthreads();
  }

  int wave = threadIdx.x >> 5;
  int lane = threadIdx.x & 31;
  int tile = blockIdx.x * 8 + wave;
  int row0 = tile * 16;                // may exceed nRows: stores self-mask
  int half = lane >> 4;
  int rl   = lane & 15;
  long long rowA = row0 + rl;
  if (rowA >= nRows) rowA = nRows - 1; // clamp loads, keep EXEC all-1s

  float inv = 0.0f;
  if constexpr (NEIGH) inv = __builtin_amdgcn_rcpf(fmaxf(Cnt[rowA], 1.0f));

  v8f acc[NT];
#pragma unroll
  for (int t = 0; t < NT; ++t) {
    int col = t * 16 + rl;
    float bv = (col < outCols) ? bias[col] : 0.0f;
#pragma unroll
    for (int r = 0; r < 8; ++r) acc[t][r] = bv;
  }

#pragma unroll
  for (int kcIdx = 0; kcIdx < 2; ++kcIdx) {
    int kc = kcIdx * 32;
    v16h aS = load_a_frag(Hs, rowA, kc, half, 1.0f);
    v16h aN = aS;
    if constexpr (NEIGH) aN = load_a_frag(Agg, rowA, kc, half, inv);
#pragma unroll
    for (int t = 0; t < NT; ++t) {
      int fidx = ((kcIdx * NT + t) * 32 + lane) * 16;
      v16h bS = *(const v16h*)(sBs + fidx);
      acc[t] = __builtin_amdgcn_wmma_f32_16x16x32_f16(
          false, aS, false, bS, (short)0, acc[t], false, false);
      if constexpr (NEIGH) {
        v16h bN = *(const v16h*)(sBn + fidx);
        acc[t] = __builtin_amdgcn_wmma_f32_16x16x32_f16(
            false, aN, false, bN, (short)0, acc[t], false, false);
      }
    }
  }

  // D layout: VGPR r -> row = row0 + r + 8*half ; col = t*16 + (lane&15)
#pragma unroll
  for (int t = 0; t < NT; ++t) {
    int col = t * 16 + rl;
    if (col >= outCols) continue;
#pragma unroll
    for (int r = 0; r < 8; ++r) {
      int row = row0 + r + half * 8;
      if (row >= nRows) continue;
      float v = acc[t][r];
      if constexpr (RELU) v = fmaxf(v, 0.0f);
      Out[(long long)row * outCols + col] = v;
    }
  }
}

// ---------------------------------------------------------------------------
// log_softmax over 40 classes (one thread per row)
// ---------------------------------------------------------------------------
__global__ void log_softmax40(const float* __restrict__ logits,
                              float* __restrict__ out, long long n) {
  long long i = (long long)blockIdx.x * blockDim.x + threadIdx.x;
  if (i >= n) return;
  const float* p = logits + i * 40;
  float m = p[0];
  for (int c = 1; c < 40; ++c) m = fmaxf(m, p[c]);
  float s = 0.0f;
  for (int c = 0; c < 40; ++c) s += expf(p[c] - m);
  float lse = m + logf(s);
  float* o = out + i * 40;
  for (int c = 0; c < 40; ++c) o[c] = p[c] - lse;
}

// ---------------------------------------------------------------------------
// Host orchestration
// ---------------------------------------------------------------------------
static inline unsigned int nblk(long long work, int bs) {
  return (unsigned int)((work + bs - 1) / bs);
}

extern "C" void kernel_launch(void* const* d_in, const int* in_sizes, int n_in,
                              void* d_out, int out_size, void* d_ws, size_t ws_size,
                              hipStream_t stream) {
  const float* x    = (const float*)d_in[0];
  const int*   n_id = (const int*)d_in[1];
  const int*   edge = (const int*)d_in[2];
  const float* W1s  = (const float*)d_in[3];
  const float* W1n  = (const float*)d_in[4];
  const float* b1   = (const float*)d_in[5];
  const float* W2s  = (const float*)d_in[6];
  const float* W2n  = (const float*)d_in[7];
  const float* b2   = (const float*)d_in[8];
  const float* Wl1  = (const float*)d_in[9];
  const float* bl1  = (const float*)d_in[10];
  const float* Wl2  = (const float*)d_in[11];
  const float* bl2  = (const float*)d_in[12];

  long long n = in_sizes[1];
  long long e = in_sizes[2] / 2;
  const int* src = edge;
  const int* dst = edge + e;

  // workspace layout (256B aligned slices)
  char* w = (char*)d_ws;
  size_t off = 0;
  auto alloc = [&](size_t bytes) -> void* {
    void* p = w + off;
    off = (off + bytes + 255) & ~(size_t)255;
    return p;
  };
  float* A   = (float*)alloc((size_t)n * 64 * 4);   // h ping
  float* B   = (float*)alloc((size_t)n * 64 * 4);   // h pong
  float* AGG = (float*)alloc((size_t)n * 64 * 4);   // aggregate / logits reuse
  float* CNT = (float*)alloc((size_t)n * 4);
  _Float16* pk = (_Float16*)alloc(6 * 4096 * sizeof(_Float16));
  _Float16* pW1s = pk + 0 * 4096;
  _Float16* pW1n = pk + 1 * 4096;
  _Float16* pW2s = pk + 2 * 4096;
  _Float16* pW2n = pk + 3 * 4096;
  _Float16* pWl1 = pk + 4 * 4096;
  _Float16* pWl2 = pk + 5 * 4096;
  float* LOGITS = AGG;   // AGG dead after conv2 GEMM; n*40 < n*64

  const int BS = 256;

  // pack fp16 weight fragments
  pack_weight<<<nblk(4096, BS), BS, 0, stream>>>(W1s, pW1s, 64, 4);
  pack_weight<<<nblk(4096, BS), BS, 0, stream>>>(W1n, pW1n, 64, 4);
  pack_weight<<<nblk(4096, BS), BS, 0, stream>>>(W2s, pW2s, 64, 4);
  pack_weight<<<nblk(4096, BS), BS, 0, stream>>>(W2n, pW2n, 64, 4);
  pack_weight<<<nblk(4096, BS), BS, 0, stream>>>(Wl1, pWl1, 64, 4);
  pack_weight<<<nblk(3072, BS), BS, 0, stream>>>(Wl2, pWl2, 40, 3);

  // degree counts (shared by both convs)
  zero_f32<<<nblk(n, BS), BS, 0, stream>>>(CNT, n);
  edge_count<<<nblk(e, BS), BS, 0, stream>>>(dst, CNT, e);

  // h0 = x[n_id]
  gather_rows<<<nblk(n * 16, BS), BS, 0, stream>>>((const float4*)x, n_id,
                                                   (float4*)A, n);

  unsigned int gemmBlocks = nblk((n + 15) / 16, 8);

  // conv1: B = relu(A@W1s + mean(A)@W1n + b1)
  zero_f32<<<nblk(n * 64, BS), BS, 0, stream>>>(AGG, n * 64);
  edge_scatter<<<nblk(e * 16, BS), BS, 0, stream>>>(src, dst, (const float4*)A,
                                                    AGG, e);
  sage_gemm<4, true, true><<<gemmBlocks, BS, 0, stream>>>(
      A, AGG, CNT, pW1s, pW1n, b1, B, (int)n, 64);

  // conv2: A = relu(B@W2s + mean(B)@W2n + b2)
  zero_f32<<<nblk(n * 64, BS), BS, 0, stream>>>(AGG, n * 64);
  edge_scatter<<<nblk(e * 16, BS), BS, 0, stream>>>(src, dst, (const float4*)B,
                                                    AGG, e);
  sage_gemm<4, true, true><<<gemmBlocks, BS, 0, stream>>>(
      B, AGG, CNT, pW2s, pW2n, b2, A, (int)n, 64);

  // mlp1: B = relu(A@Wl1 + bl1)
  sage_gemm<4, false, true><<<gemmBlocks, BS, 0, stream>>>(
      A, nullptr, nullptr, pWl1, nullptr, bl1, B, (int)n, 64);

  // mlp2: LOGITS = B@Wl2 + bl2   (40 cols, 3 tiles)
  sage_gemm<3, false, false><<<gemmBlocks, BS, 0, stream>>>(
      B, nullptr, nullptr, pWl2, nullptr, bl2, LOGITS, (int)n, 40);

  // log_softmax -> d_out
  log_softmax40<<<nblk(n, BS), BS, 0, stream>>>(LOGITS, (float*)d_out, n);
}
